// meta_solver_gru_26620207300634
// MI455X (gfx1250) — matrix operand order repaired and checked by reading, not compiled, not run on hardware
//
#include <hip/hip_runtime.h>

#define NN    1024
#define GATES 96

typedef __attribute__((ext_vector_type(16))) _Float16 v16h;
typedef __attribute__((ext_vector_type(8)))  float    v8f;
typedef __attribute__((ext_vector_type(4)))  float    v4f;

__device__ __forceinline__ v8f load8(const float* __restrict__ p) {
  const v4f* q = (const v4f*)p;
  v4f a = q[0], b = q[1];
  v8f r;
  r[0]=a[0]; r[1]=a[1]; r[2]=a[2]; r[3]=a[3];
  r[4]=b[0]; r[5]=b[1]; r[6]=b[2]; r[7]=b[3];
  return r;
}

__device__ __forceinline__ void store8(float* __restrict__ p, v8f x) {
  v4f a, b;
  a[0]=x[0]; a[1]=x[1]; a[2]=x[2]; a[3]=x[3];
  b[0]=x[4]; b[1]=x[5]; b[2]=x[6]; b[3]=x[7];
  v4f* q = (v4f*)p;
  q[0] = a; q[1] = b;
}

__device__ __forceinline__ float fast_sigmoid(float x) {
  float e = __builtin_amdgcn_exp2f(-1.44269504088896340736f * x);
  return __builtin_amdgcn_rcpf(1.0f + e);
}
__device__ __forceinline__ float fast_tanh(float x) {
  // tanh(x) = (1 - e) / (1 + e),  e = exp(-2x)
  float e = __builtin_amdgcn_exp2f(-2.88539008177792681472f * x);
  return 1.0f - 2.0f * e * __builtin_amdgcn_rcpf(1.0f + e);
}

// ---------------------------------------------------------------------------
// 1) Transpose x (1024x1024) so scan step t reads contiguous xT[t*1024 + b].
// ---------------------------------------------------------------------------
__global__ __launch_bounds__(256)
void transpose_kernel(const float* __restrict__ x, float* __restrict__ xT) {
  __shared__ float tile[32][33];
  const int tx = threadIdx.x, ty = threadIdx.y;
  const int bx = blockIdx.x * 32, by = blockIdx.y * 32;
#pragma unroll
  for (int i = 0; i < 32; i += 8)
    tile[ty + i][tx] = x[(by + ty + i) * NN + bx + tx];
  __syncthreads();
#pragma unroll
  for (int i = 0; i < 32; i += 8)
    xT[(bx + ty + i) * NN + by + tx] = tile[tx][ty + i];
}

// ---------------------------------------------------------------------------
// 2) Fold l1 (scalar->32) + GRU1 input projection + biases into 96-vectors:
//    gi1[b,t,g] = x[b,t]*A96[g] + {Bc96|Bn96}[g]
//    A96  = Wih @ l1_w
//    Bn96 = Wih @ l1_b + bih                (n-gate input part; bhh stays inside r*)
//    Bc96 = Bn96 + bhh                      (r/z gates: both biases folded)
// ---------------------------------------------------------------------------
__global__ void prep_kernel(const float* __restrict__ Wih, const float* __restrict__ bih,
                            const float* __restrict__ bhh, const float* __restrict__ l1w,
                            const float* __restrict__ l1b,
                            float* __restrict__ A96o, float* __restrict__ Bc96o,
                            float* __restrict__ Bn96o) {
  const int g = threadIdx.x;
  if (g >= GATES) return;
  float a = 0.f, b = 0.f;
#pragma unroll
  for (int i = 0; i < 32; ++i) {
    const float w = Wih[g * 32 + i];
    a += w * l1w[i];
    b += w * l1b[i];
  }
  A96o[g] = a;
  const float bn = b + bih[g];
  Bn96o[g] = bn;
  Bc96o[g] = bn + bhh[g];
}

// ---------------------------------------------------------------------------
// 3) Precompute GRU2 input gates: gi2[t,g] = Wih2[g,:]·h1[t,:] + bih2 (+bhh2 for r/z)
// ---------------------------------------------------------------------------
__global__ void gi2_kernel(const float* __restrict__ h1, const float* __restrict__ Wih2,
                           const float* __restrict__ bih2, const float* __restrict__ bhh2,
                           float* __restrict__ gi2) {
  const int t = blockIdx.x, g = threadIdx.x;
  if (g >= GATES) return;
  float acc = bih2[g] + (g < 64 ? bhh2[g] : 0.f);
  const float* hr = h1 + t * 32;
  const float* wr = Wih2 + g * 32;
#pragma unroll
  for (int i = 0; i < 32; ++i) acc += wr[i] * hr[i];
  gi2[t * GATES + g] = acc;
}

// ---------------------------------------------------------------------------
// 4) GRU scan on WMMA.  One wave owns a 16-row tile of the batch.
//    Transposed product: D_c(16x16) = A_c(16x32 gates x hidden) x h^T(32x16) + C
//    Gate rows permuted:  g(c,m) = 32*(c/2) + 8*(c%2) + (m%8) + 16*(m/8)
//    so that per-lane gate-math output is already in the WMMA f16 B-operand
//    layout for the next step (lane l<16: hidden 0..15 of row l; lane>=16:
//    hidden 16..31).  Chunks 0,1: r; 2,3: z; 4,5: n (C preloaded with bhh_n).
//    MODE 0: GRU1 (scalar-input trick, 64 blocks x 16 rows)
//    MODE 1: GRU2 (batch 1, gi2 precomputed, only column 0 meaningful;
//            all columns compute identically so no garbage propagates)
// ---------------------------------------------------------------------------
template <int MODE>
__global__ __launch_bounds__(32)
void gru_scan_kernel(const float* __restrict__ xT,
                     const float* __restrict__ gi2,
                     const float* __restrict__ Whh,
                     const float* __restrict__ A96,
                     const float* __restrict__ Bc96,
                     const float* __restrict__ Bn96,
                     const float* __restrict__ bhh,
                     float* __restrict__ hout,
                     int T) {
  const int l = threadIdx.x & 31;
  const int u = l >> 4;        // lane half
  const int col = l & 15;      // batch row within tile (and A-operand row M)
  const int base = blockIdx.x << 4;

  // Load constant A operand: permuted Whh rows in the 16-bit A layout
  // (contiguous K order: lane half u holds K 16u..16u+15).
  v16h A[6];
#pragma unroll
  for (int c = 0; c < 6; ++c) {
    const int g = 32 * (c >> 1) + 8 * (c & 1) + (col & 7) + 16 * (col >> 3);
    const float* wr = Whh + g * 32;
#pragma unroll
    for (int e = 0; e < 16; ++e) {
      const int k = e + 16 * u;
      A[c][e] = (_Float16)wr[k];
    }
  }

  v8f Ag[6], Bc[4], Bn[2], Bh[2];
  if (MODE == 0) {
#pragma unroll
    for (int c = 0; c < 6; ++c)
      Ag[c] = load8(A96 + 32 * (c >> 1) + 8 * (c & 1) + 16 * u);
#pragma unroll
    for (int c = 0; c < 4; ++c)
      Bc[c] = load8(Bc96 + 32 * (c >> 1) + 8 * (c & 1) + 16 * u);
    Bn[0] = load8(Bn96 + 64 + 16 * u);
    Bn[1] = load8(Bn96 + 72 + 16 * u);
  }
  Bh[0] = load8(bhh + 64 + 16 * u);   // bhh for n-gate chunks (stays inside r*)
  Bh[1] = load8(bhh + 72 + 16 * u);

  v8f h0  = {0, 0, 0, 0, 0, 0, 0, 0};   // hidden 16u + 0..7 per lane (f32 state)
  v8f h1s = {0, 0, 0, 0, 0, 0, 0, 0};   // hidden 16u + 8..15 per lane
  v16h hB = {0, 0, 0, 0, 0, 0, 0, 0, 0, 0, 0, 0, 0, 0, 0, 0};  // f16 B operand

  for (int t = 0; t < T; ++t) {
    v8f C0, C1, C2, C3, gin0, gin1;
    if (MODE == 0) {
      const float xs = xT[t * NN + base + col];
      C0 = xs * Ag[0] + Bc[0];
      C1 = xs * Ag[1] + Bc[1];
      C2 = xs * Ag[2] + Bc[2];
      C3 = xs * Ag[3] + Bc[3];
      gin0 = xs * Ag[4] + Bn[0];
      gin1 = xs * Ag[5] + Bn[1];
    } else {
      const float* gp = gi2 + t * GATES;
      C0 = load8(gp + 0 + 16 * u);
      C1 = load8(gp + 8 + 16 * u);
      C2 = load8(gp + 32 + 16 * u);
      C3 = load8(gp + 40 + 16 * u);
      gin0 = load8(gp + 64 + 16 * u);
      gin1 = load8(gp + 72 + 16 * u);
    }

    const v8f D0 = __builtin_amdgcn_wmma_f32_16x16x32_f16(false, A[0], false, hB, (short)0, C0, false, false);
    const v8f D1 = __builtin_amdgcn_wmma_f32_16x16x32_f16(false, A[1], false, hB, (short)0, C1, false, false);
    const v8f D2 = __builtin_amdgcn_wmma_f32_16x16x32_f16(false, A[2], false, hB, (short)0, C2, false, false);
    const v8f D3 = __builtin_amdgcn_wmma_f32_16x16x32_f16(false, A[3], false, hB, (short)0, C3, false, false);
    const v8f D4 = __builtin_amdgcn_wmma_f32_16x16x32_f16(false, A[4], false, hB, (short)0, Bh[0], false, false);
    const v8f D5 = __builtin_amdgcn_wmma_f32_16x16x32_f16(false, A[5], false, hB, (short)0, Bh[1], false, false);

    v16h hBn;
#pragma unroll
    for (int v = 0; v < 8; ++v) {
      // hidden j = 16u + v
      const float r0 = fast_sigmoid(D0[v]);
      const float z0 = fast_sigmoid(D2[v]);
      const float n0 = fast_tanh(gin0[v] + r0 * D4[v]);
      const float hv0 = n0 + z0 * (h0[v] - n0);
      h0[v] = hv0;
      hBn[v] = (_Float16)hv0;
      // hidden j = 16u + 8 + v
      const float r1 = fast_sigmoid(D1[v]);
      const float z1 = fast_sigmoid(D3[v]);
      const float n1 = fast_tanh(gin1[v] + r1 * D5[v]);
      const float hv1 = n1 + z1 * (h1s[v] - n1);
      h1s[v] = hv1;
      hBn[8 + v] = (_Float16)hv1;
    }
    hB = hBn;  // already in next step's B-operand layout
  }

  if (MODE == 0) {
    float* o = hout + (base + col) * 32 + 16 * u;  // hidden 16u..16u+15 contiguous
    store8(o, h0);
    store8(o + 8, h1s);
  } else {
    if (col == 0) {
      float* o = hout + 16 * u;
      store8(o, h0);
      store8(o + 8, h1s);
    }
  }
}

// ---------------------------------------------------------------------------
// 5) MLP head + softmax over 1024 rows (one 1024-thread block).
// ---------------------------------------------------------------------------
__global__ __launch_bounds__(1024)
void head_kernel(const float* __restrict__ h1, const float* __restrict__ h2,
                 const float* __restrict__ f1W, const float* __restrict__ f1b,
                 const float* __restrict__ f2W, const float* __restrict__ f2b,
                 const float* __restrict__ f3W, const float* __restrict__ f3b,
                 float* __restrict__ out) {
  __shared__ float smax[32], ssum[32];
  const int r = threadIdx.x;

  float xl[32], xg[32];
#pragma unroll
  for (int i = 0; i < 32; ++i) {
    xl[i] = h1[r * 32 + i];
    xg[i] = h2[i];
  }
  float y1[32];
#pragma unroll
  for (int o = 0; o < 32; ++o) {
    float acc = f1b[o];
    const float* w = f1W + o * 64;
#pragma unroll
    for (int i = 0; i < 32; ++i) acc += w[i] * xl[i];
#pragma unroll
    for (int i = 0; i < 32; ++i) acc += w[32 + i] * xg[i];
    y1[o] = fmaxf(acc, 0.f);
  }
  float s = f3b[0];
#pragma unroll
  for (int o = 0; o < 32; ++o) {
    float acc = f2b[o];
    const float* w = f2W + o * 32;
#pragma unroll
    for (int i = 0; i < 32; ++i) acc += w[i] * y1[i];
    s += f3W[o] * fmaxf(acc, 0.f);
  }

  // block-wide max
  float m = s;
#pragma unroll
  for (int off = 16; off > 0; off >>= 1) m = fmaxf(m, __shfl_xor(m, off, 32));
  if ((r & 31) == 0) smax[r >> 5] = m;
  __syncthreads();
  if (r < 32) {
    float mm = smax[r];
#pragma unroll
    for (int off = 16; off > 0; off >>= 1) mm = fmaxf(mm, __shfl_xor(mm, off, 32));
    smax[r] = mm;
  }
  __syncthreads();
  const float M = smax[0];
  const float e = __builtin_amdgcn_exp2f(1.44269504088896340736f * (s - M));
  // block-wide sum
  float sum = e;
#pragma unroll
  for (int off = 16; off > 0; off >>= 1) sum += __shfl_xor(sum, off, 32);
  if ((r & 31) == 0) ssum[r >> 5] = sum;
  __syncthreads();
  if (r < 32) {
    float ss = ssum[r];
#pragma unroll
    for (int off = 16; off > 0; off >>= 1) ss += __shfl_xor(ss, off, 32);
    ssum[r] = ss;
  }
  __syncthreads();
  out[r] = e * __builtin_amdgcn_rcpf(ssum[0]);
}

// ---------------------------------------------------------------------------
extern "C" void kernel_launch(void* const* d_in, const int* in_sizes, int n_in,
                              void* d_out, int out_size, void* d_ws, size_t ws_size,
                              hipStream_t stream) {
  const float* x     = (const float*)d_in[0];
  const float* l1W   = (const float*)d_in[1];
  const float* l1b   = (const float*)d_in[2];
  const float* g1Wih = (const float*)d_in[3];
  const float* g1Whh = (const float*)d_in[4];
  const float* g1bih = (const float*)d_in[5];
  const float* g1bhh = (const float*)d_in[6];
  const float* g2Wih = (const float*)d_in[7];
  const float* g2Whh = (const float*)d_in[8];
  const float* g2bih = (const float*)d_in[9];
  const float* g2bhh = (const float*)d_in[10];
  const float* f1W   = (const float*)d_in[11];
  const float* f1b   = (const float*)d_in[12];
  const float* f2W   = (const float*)d_in[13];
  const float* f2b   = (const float*)d_in[14];
  const float* f3W   = (const float*)d_in[15];
  const float* f3b   = (const float*)d_in[16];
  float* out = (float*)d_out;

  // Workspace layout (floats), ~4.7 MB total, all 16B-aligned sub-buffers.
  float* ws   = (float*)d_ws;
  float* xT   = ws;                  // 1024*1024
  float* h1   = xT + NN * NN;        // 1024*32
  float* gi2  = h1 + NN * 32;        // 1024*96
  float* h2   = gi2 + NN * GATES;    // 32
  float* A96  = h2 + 32;             // 96
  float* Bc96 = A96 + 96;            // 96
  float* Bn96 = Bc96 + 96;           // 96

  transpose_kernel<<<dim3(32, 32), dim3(32, 8), 0, stream>>>(x, xT);
  prep_kernel<<<1, 96, 0, stream>>>(g1Wih, g1bih, g1bhh, l1W, l1b, A96, Bc96, Bn96);
  gru_scan_kernel<0><<<64, 32, 0, stream>>>(xT, nullptr, g1Whh, A96, Bc96, Bn96,
                                            g1bhh, h1, NN);
  gi2_kernel<<<NN, 96, 0, stream>>>(h1, g2Wih, g2bih, g2bhh, gi2);
  gru_scan_kernel<1><<<1, 32, 0, stream>>>(nullptr, gi2, g2Whh, nullptr, nullptr,
                                           nullptr, g2bhh, h2, NN);
  head_kernel<<<1, 1024, 0, stream>>>(h1, h2, f1W, f1b, f2W, f2b, f3W, f3b, out);
}
